// ReasonerAttention_2233382994338
// MI455X (gfx1250) — matrix operand order repaired and checked by reading, not compiled, not run on hardware
//
#include <hip/hip_runtime.h>
#include <hip/hip_bf16.h>

typedef __bf16 bf16_t;
typedef __attribute__((ext_vector_type(16))) bf16_t v16bf;
typedef __attribute__((ext_vector_type(8)))  bf16_t v8bf;
typedef __attribute__((ext_vector_type(8)))  float  v8f;

// Problem constants (match reference)
constexpr int B_  = 2;
constexpr int L_  = 2048;
constexpr int HID_= 2048;
constexpr int H_  = 32;
constexpr int KVH_= 8;
constexpr int D_  = 64;
constexpr int BL_ = B_ * L_;          // 4096
constexpr int HD_ = H_ * D_;          // 2048
constexpr int KVD_= KVH_ * D_;        // 512

// ---------------------------------------------------------------------------
// WMMA helpers (CDNA5: V_WMMA_F32_16X16X32_BF16, wave32)
// ---------------------------------------------------------------------------
__device__ __forceinline__ v8f wmma_bf16(v16bf a, v16bf b, v8f c) {
  // (neg_a, A, neg_b, B, c_mod, C, reuse_a, reuse_b)
  return __builtin_amdgcn_wmma_f32_16x16x32_bf16(false, a, false, b,
                                                 (short)0, c, false, false);
}

// A-fragment, 16x32 bf16 (M x K), row-major source with leading dim `lda`.
// Contiguous-k layout used here: lanes 0-15 -> M=lane, hold K=0..15;
//             lanes 16-31 -> M=lane-16, hold K=16..31.
__device__ __forceinline__ v16bf load_a_frag(const bf16_t* base, int lda) {
  int lane = threadIdx.x & 31;
  int hi8  = (lane & 16) ? 16 : 0;
  const bf16_t* p = base + (size_t)(lane & 15) * lda + hi8;
  v8bf lo = *(const v8bf*)p;          // K = hi8 .. hi8+7
  v8bf hh = *(const v8bf*)(p + 8);    // K = hi8+8 .. hi8+15
  return __builtin_shufflevector(lo, hh, 0,1,2,3,4,5,6,7,8,9,10,11,12,13,14,15);
}

// B-fragment, 32x16 bf16 (K x N). Source: column n's K-run is contiguous with
// leading dim `ldb` between columns (i.e. row-major [N, K] layout).
// Contiguous-k layout used here: lanes 0-15 -> N=lane, K=0..15; lanes 16-31 -> N=lane-16, K=16..31.
__device__ __forceinline__ v16bf load_b_frag(const bf16_t* base, int ldb) {
  int lane = threadIdx.x & 31;
  const bf16_t* p = base + (size_t)(lane & 15) * ldb + ((lane & 16) ? 16 : 0);
  return *(const v16bf*)p;
}

// ---------------------------------------------------------------------------
// CDNA5 async copy global -> LDS (GLOBAL_LOAD_ASYNC_TO_LDS_B128, ASYNCcnt)
// VDST VGPR carries the per-lane LDS byte offset (ISA 15.18.3 op 98).
// ---------------------------------------------------------------------------
__device__ __forceinline__ unsigned lds_off_u32(void* p) {
  return (unsigned)(unsigned long long)(__attribute__((address_space(3))) void*)p;
}

__device__ __forceinline__ void async_copy_b128(const bf16_t* gsrc, unsigned ldsoff) {
  unsigned long long ga = (unsigned long long)gsrc;
  asm volatile("global_load_async_to_lds_b128 %0, %1, off"
               :: "v"(ldsoff), "v"(ga) : "memory");
}

__device__ __forceinline__ void wait_asynccnt0() {
  asm volatile("s_wait_asynccnt 0x0" ::: "memory");
}

// ---------------------------------------------------------------------------
// fp32 -> bf16 casts
// ---------------------------------------------------------------------------
__global__ void cast_bf16_kernel(const float* __restrict__ in,
                                 bf16_t* __restrict__ out, int n) {
  for (int i = blockIdx.x * blockDim.x + threadIdx.x; i < n;
       i += gridDim.x * blockDim.x)
    out[i] = (bf16_t)in[i];
}

// ---------------------------------------------------------------------------
// GEMM: C[M,N] (f32) = A[M,K](bf16, row-major) * W[N,K](bf16, row-major)^T
// 256 threads = 8 waves (2x4). Block tile 128x128, K-step 32.
// A/B tiles staged in LDS by GLOBAL_LOAD_ASYNC_TO_LDS_B128, double-buffered.
// ---------------------------------------------------------------------------
__global__ __launch_bounds__(256)
void gemm_bf16_lds_kernel(const bf16_t* __restrict__ A, const bf16_t* __restrict__ W,
                          float* __restrict__ C, int M, int N, int K) {
  __shared__ __align__(32) bf16_t Atile[2][128 * 32];   // 8 KB per buffer
  __shared__ __align__(32) bf16_t Btile[2][128 * 32];

  int tid  = threadIdx.x;
  int lane = tid & 31;
  int wave = tid >> 5;
  int wm = wave >> 2;           // 0..1 -> 64-row slab
  int wn = wave & 3;            // 0..3 -> 32-col slab
  int tMb = blockIdx.y * 128;
  int tNb = blockIdx.x * 128;

  // Each thread stages one 16-element (32 B) segment per tile per K-step.
  int crow = tid >> 1;              // 0..127
  int ccol = (tid & 1) * 16;        // 0 or 16
  const bf16_t* Ag = A + (size_t)(tMb + crow) * K + ccol;
  const bf16_t* Bg = W + (size_t)(tNb + crow) * K + ccol;
  unsigned la[2], lb[2];
  la[0] = lds_off_u32(&Atile[0][crow * 32 + ccol]);
  la[1] = lds_off_u32(&Atile[1][crow * 32 + ccol]);
  lb[0] = lds_off_u32(&Btile[0][crow * 32 + ccol]);
  lb[1] = lds_off_u32(&Btile[1][crow * 32 + ccol]);

  v8f acc[4][2] = {};

  // Prologue: stage first K-tile into buffer 0.
  async_copy_b128(Ag, la[0]);
  async_copy_b128(Bg, lb[0]);

  int p = 0;
  for (int k0 = 0; k0 < K; k0 += 32) {
    wait_asynccnt0();               // my async copies for buffer p landed
    __syncthreads();                // everyone's landed; prev buffer fully read
    if (k0 + 32 < K) {              // overlap next tile with compute
      async_copy_b128(Ag + k0 + 32, la[p ^ 1]);
      async_copy_b128(Bg + k0 + 32, lb[p ^ 1]);
    }
    v16bf bfr0 = load_b_frag(&Btile[p][(wn * 32 +  0) * 32], 32);
    v16bf bfr1 = load_b_frag(&Btile[p][(wn * 32 + 16) * 32], 32);
#pragma unroll
    for (int f = 0; f < 4; ++f) {
      v16bf afr = load_a_frag(&Atile[p][(wm * 64 + f * 16) * 32], 32);
      acc[f][0] = wmma_bf16(afr, bfr0, acc[f][0]);
      acc[f][1] = wmma_bf16(afr, bfr1, acc[f][1]);
    }
    p ^= 1;
  }

  int colr = lane & 15;
  int rofs = (lane & 16) ? 8 : 0;
#pragma unroll
  for (int f = 0; f < 4; ++f)
#pragma unroll
    for (int j = 0; j < 2; ++j)
#pragma unroll
      for (int i = 0; i < 8; ++i) {
        int r  = tMb + wm * 64 + f * 16 + rofs + i;
        int cc = tNb + wn * 32 + j * 16 + colr;
        C[(size_t)r * N + cc] = acc[f][j][i];
      }
}

// ---------------------------------------------------------------------------
// RoPE: read f32 [B, L, heads*D] (GEMM output), rotate pairs, write bf16
// transposed to [B, heads, L, D]. `oscale` folds the attention 1/sqrt(D)
// into Q at conversion time (1.0 for K).
// ---------------------------------------------------------------------------
__global__ void rope_kernel(const float* __restrict__ t,
                            const float* __restrict__ cs,
                            const float* __restrict__ sn,
                            bf16_t* __restrict__ o, int heads, float oscale,
                            int total) {
  int idx = blockIdx.x * blockDim.x + threadIdx.x;
  if (idx >= total) return;
  int j   = idx & 31;            // D/2 = 32
  int tmp = idx >> 5;
  int h   = tmp % heads; tmp /= heads;
  int l   = tmp % L_;
  int b   = tmp / L_;
  size_t ibase = ((size_t)(b * L_ + l) * heads + h) * D_ + 2 * j;
  float x1 = t[ibase], x2 = t[ibase + 1];
  float c = cs[l * 32 + j], s = sn[l * 32 + j];
  size_t obase = (((size_t)b * heads + h) * L_ + l) * D_ + 2 * j;
  o[obase]     = (bf16_t)((x1 * c - x2 * s) * oscale);
  o[obase + 1] = (bf16_t)((x1 * s + x2 * c) * oscale);
}

// V: cast + transpose [B, L, KVH*D] f32 -> [B, KVH, L, D] bf16
__global__ void vcast_kernel(const float* __restrict__ vf,
                             bf16_t* __restrict__ vb, int total) {
  int idx = blockIdx.x * blockDim.x + threadIdx.x;
  if (idx >= total) return;
  int d   = idx & 63;
  int tmp = idx >> 6;
  int h   = tmp % KVH_; tmp /= KVH_;
  int l   = tmp % L_;
  int b   = tmp / L_;
  vb[(((size_t)b * KVH_ + h) * L_ + l) * D_ + d] =
      (bf16_t)vf[((size_t)(b * L_ + l) * KVH_ + h) * D_ + d];
}

// ---------------------------------------------------------------------------
// Flash attention (causal, GQA 4:1).
// Block = 128 threads = 4 waves sharing one (b,h) and a 64-row Q super-tile;
// each wave owns 16 Q rows. K/V 32x64 tiles staged cooperatively in LDS via
// GLOBAL_LOAD_ASYNC_TO_LDS_B128, double-buffered; loop count is block-uniform
// (lower waves see fully-masked tails, nulled by the online softmax).
// Q,K,V: bf16 [B, heads, L, D] (Q pre-scaled by 1/sqrt(D)).
// Output: bf16 [B, L, H*D].
// ---------------------------------------------------------------------------
__global__ __launch_bounds__(128)
void flash_attn_kernel(const bf16_t* __restrict__ Qg,
                       const bf16_t* __restrict__ Kg,
                       const bf16_t* __restrict__ Vg,
                       bf16_t* __restrict__ Og) {
  __shared__ __align__(32) bf16_t Kt[2][32 * 64];     // 4 KB per buffer
  __shared__ __align__(32) bf16_t Vt[2][32 * 64];
  __shared__ __align__(32) bf16_t Plds[4][16 * 32];   // per-wave P tile

  int tid  = threadIdx.x;
  int lane = tid & 31;
  int wave = tid >> 5;
  int b = blockIdx.z, h = blockIdx.y;
  int qsuper = blockIdx.x * 64;
  int q0 = qsuper + wave * 16;
  int kvh = h >> 2;                                   // N_REP = 4

  const bf16_t* Q  = Qg + (((size_t)b * H_   + h  ) * L_) * D_;
  const bf16_t* Kp = Kg + (((size_t)b * KVH_ + kvh) * L_) * D_;
  const bf16_t* Vp = Vg + (((size_t)b * KVH_ + kvh) * L_) * D_;

  // Cooperative staging map: 128 threads x 32 B cover one 32x64 bf16 tile.
  int srow = tid >> 2;               // 0..31 (key row within tile)
  int scol = (tid & 3) * 16;         // 0,16,32,48 (d column)
  const bf16_t* Kg2 = Kp + (size_t)srow * D_ + scol;
  const bf16_t* Vg2 = Vp + (size_t)srow * D_ + scol;
  unsigned lk[2], lv[2];
  lk[0] = lds_off_u32(&Kt[0][srow * 64 + scol]);
  lk[1] = lds_off_u32(&Kt[1][srow * 64 + scol]);
  lv[0] = lds_off_u32(&Vt[0][srow * 64 + scol]);
  lv[1] = lds_off_u32(&Vt[1][srow * 64 + scol]);

  v16bf qa0 = load_a_frag(Q + (size_t)q0 * D_ +  0, D_);
  v16bf qa1 = load_a_frag(Q + (size_t)q0 * D_ + 32, D_);

  float mrow[8], lrow[8];
  v8f o[4] = {};
#pragma unroll
  for (int i = 0; i < 8; ++i) { mrow[i] = -3.0e38f; lrow[i] = 0.0f; }

  int colr = lane & 15;
  int rofs = (lane & 16) ? 8 : 0;
  int kvofs = (lane & 16) ? 16 : 0;

  int kend = qsuper + 32;            // last key tile start (covers qsuper+63)

  // Prologue: stage first K/V tile into buffer 0.
  async_copy_b128(Kg2, lk[0]);
  async_copy_b128(Vg2, lv[0]);

  int p = 0;
  for (int k0 = 0; k0 <= kend; k0 += 32) {
    wait_asynccnt0();                // my async copies for buffer p landed
    __syncthreads();                 // everyone's landed; prev buffer consumed
    if (k0 + 32 <= kend) {           // overlap next K/V tile with compute
      async_copy_b128(Kg2 + (size_t)(k0 + 32) * D_, lk[p ^ 1]);
      async_copy_b128(Vg2 + (size_t)(k0 + 32) * D_, lv[p ^ 1]);
    }

    // S = Q * K^T for 32 keys as two 16-key halves, D split 32+32 (from LDS)
    v16bf kf00 = load_b_frag(&Kt[p][ 0 * 64 +  0], 64);
    v16bf kf01 = load_b_frag(&Kt[p][ 0 * 64 + 32], 64);
    v16bf kf10 = load_b_frag(&Kt[p][16 * 64 +  0], 64);
    v16bf kf11 = load_b_frag(&Kt[p][16 * 64 + 32], 64);
    v8f s0 = {}; s0 = wmma_bf16(qa0, kf00, s0); s0 = wmma_bf16(qa1, kf01, s0);
    v8f s1 = {}; s1 = wmma_bf16(qa0, kf10, s1); s1 = wmma_bf16(qa1, kf11, s1);

    // Online softmax. C layout: element i -> row q0+rofs+i, lane -> column.
#pragma unroll
    for (int i = 0; i < 8; ++i) {
      int row = q0 + rofs + i;
      float x0 = s0[i]; if (k0 +      colr > row) x0 = -1.0e9f;
      float x1 = s1[i]; if (k0 + 16 + colr > row) x1 = -1.0e9f;
      float mx = fmaxf(x0, x1);
      mx = fmaxf(mx, __shfl_xor(mx, 1, 32));
      mx = fmaxf(mx, __shfl_xor(mx, 2, 32));
      mx = fmaxf(mx, __shfl_xor(mx, 4, 32));
      mx = fmaxf(mx, __shfl_xor(mx, 8, 32));
      float mn    = fmaxf(mrow[i], mx);
      float alpha = __expf(mrow[i] - mn);
      float p0 = __expf(x0 - mn);
      float p1 = __expf(x1 - mn);
      float ps = p0 + p1;
      ps += __shfl_xor(ps, 1, 32);
      ps += __shfl_xor(ps, 2, 32);
      ps += __shfl_xor(ps, 4, 32);
      ps += __shfl_xor(ps, 8, 32);
      lrow[i] = lrow[i] * alpha + ps;
      mrow[i] = mn;
      o[0][i] *= alpha; o[1][i] *= alpha; o[2][i] *= alpha; o[3][i] *= alpha;
      // Stage P (C layout) into LDS so it can be re-read in A-fragment layout
      Plds[wave][(rofs + i) * 32 +      colr] = (bf16_t)p0;
      Plds[wave][(rofs + i) * 32 + 16 + colr] = (bf16_t)p1;
    }
    asm volatile("s_wait_dscnt 0x0" ::: "memory");     // intra-wave LDS RAW

    v16bf pa = load_a_frag(&Plds[wave][0], 32);        // P as 16x32 A-matrix

    // O += P * V ; V B-fragment: lane -> column d, 16 keys per lane (LDS)
#pragma unroll
    for (int c = 0; c < 4; ++c) {
      int dcol = c * 16 + colr;
      v16bf vfrag;
#pragma unroll
      for (int j = 0; j < 16; ++j)
        vfrag[j] = Vt[p][(kvofs + j) * 64 + dcol];
      o[c] = wmma_bf16(pa, vfrag, o[c]);
    }
    p ^= 1;
  }

  // Normalize and store to [B, L, H*D] bf16 (A-matrix-ready for out-proj GEMM)
#pragma unroll
  for (int i = 0; i < 8; ++i) {
    int row = q0 + rofs + i;
    float inv = 1.0f / lrow[i];
    size_t base = ((size_t)b * L_ + row) * (size_t)HD_ + (size_t)h * D_;
#pragma unroll
    for (int c = 0; c < 4; ++c)
      Og[base + c * 16 + colr] = (bf16_t)(o[c][i] * inv);
  }
}

// ---------------------------------------------------------------------------
// Host-side orchestration
// ---------------------------------------------------------------------------
extern "C" void kernel_launch(void* const* d_in, const int* in_sizes, int n_in,
                              void* d_out, int out_size, void* d_ws, size_t ws_size,
                              hipStream_t stream) {
  (void)in_sizes; (void)n_in; (void)out_size; (void)ws_size;
  const float* x    = (const float*)d_in[0];
  const float* wq   = (const float*)d_in[1];
  const float* wk   = (const float*)d_in[2];
  const float* wv   = (const float*)d_in[3];
  const float* wo   = (const float*)d_in[4];
  const float* cosp = (const float*)d_in[5];
  const float* sinp = (const float*)d_in[6];
  // d_in[7] = mask (causal, folded into kernel), d_in[8] = start_pos (0)

  char* ws = (char*)d_ws;
  size_t off = 0;
  auto alloc = [&](size_t bytes) -> char* {
    char* p = ws + off;
    off += (bytes + 255) & ~(size_t)255;
    return p;
  };

  bf16_t* xb  = (bf16_t*)alloc((size_t)BL_ * HID_ * 2);
  bf16_t* wqb = (bf16_t*)alloc((size_t)HD_ * HID_ * 2);
  bf16_t* wkb = (bf16_t*)alloc((size_t)KVD_ * HID_ * 2);
  bf16_t* wvb = (bf16_t*)alloc((size_t)KVD_ * HID_ * 2);
  bf16_t* wob = (bf16_t*)alloc((size_t)HID_ * HD_ * 2);
  float*  qf  = (float*)alloc((size_t)BL_ * HD_ * 4);
  float*  kf  = (float*)alloc((size_t)BL_ * KVD_ * 4);
  float*  vf  = (float*)alloc((size_t)BL_ * KVD_ * 4);
  bf16_t* qb  = (bf16_t*)alloc((size_t)BL_ * HD_ * 2);
  bf16_t* kb  = (bf16_t*)alloc((size_t)BL_ * KVD_ * 2);
  bf16_t* vb  = (bf16_t*)alloc((size_t)BL_ * KVD_ * 2);
  bf16_t* attnb = (bf16_t*)qf;   // reuse: qf dead after RoPE, before attention

  // 1) casts to bf16
  cast_bf16_kernel<<<2048, 256, 0, stream>>>(x,  xb,  BL_ * HID_);
  cast_bf16_kernel<<<2048, 256, 0, stream>>>(wq, wqb, HD_ * HID_);
  cast_bf16_kernel<<<1024, 256, 0, stream>>>(wk, wkb, KVD_ * HID_);
  cast_bf16_kernel<<<1024, 256, 0, stream>>>(wv, wvb, KVD_ * HID_);
  cast_bf16_kernel<<<2048, 256, 0, stream>>>(wo, wob, HID_ * HD_);

  // 2) QKV projections (WMMA + async LDS staging)
  gemm_bf16_lds_kernel<<<dim3(HD_ / 128,  BL_ / 128), 256, 0, stream>>>(xb, wqb, qf, BL_, HD_,  HID_);
  gemm_bf16_lds_kernel<<<dim3(KVD_ / 128, BL_ / 128), 256, 0, stream>>>(xb, wkb, kf, BL_, KVD_, HID_);
  gemm_bf16_lds_kernel<<<dim3(KVD_ / 128, BL_ / 128), 256, 0, stream>>>(xb, wvb, vf, BL_, KVD_, HID_);

  // 3) RoPE + transpose to [B, heads, L, D] bf16 (Q pre-scaled by 1/sqrt(D))
  {
    int tq = B_ * L_ * H_ * (D_ / 2);
    rope_kernel<<<(tq + 255) / 256, 256, 0, stream>>>(qf, cosp, sinp, qb, H_, 0.125f, tq);
    int tk = B_ * L_ * KVH_ * (D_ / 2);
    rope_kernel<<<(tk + 255) / 256, 256, 0, stream>>>(kf, cosp, sinp, kb, KVH_, 1.0f, tk);
    int tv = B_ * L_ * KVH_ * D_;
    vcast_kernel<<<(tv + 255) / 256, 256, 0, stream>>>(vf, vb, tv);
  }

  // 4) causal GQA flash attention (WMMA + async LDS staging of K/V)
  flash_attn_kernel<<<dim3(L_ / 64, H_, B_), 128, 0, stream>>>(qb, kb, vb, attnb);

  // 5) output projection (WMMA + async LDS staging) -> f32 d_out
  gemm_bf16_lds_kernel<<<dim3(HID_ / 128, BL_ / 128), 256, 0, stream>>>(attnb, wob, (float*)d_out,
                                                                        BL_, HID_, HD_);
}
